// SelfAttention_8031588843930
// MI455X (gfx1250) — compile-verified, matrix operand order checked by reading, not run on hardware
//
#include <hip/hip_runtime.h>
#include <hip/hip_bf16.h>
#include <math.h>

// ---------------- types ----------------
typedef __bf16 bf16_t;
typedef __attribute__((ext_vector_type(16))) __bf16 v16bf;
typedef __attribute__((ext_vector_type(8)))  __bf16 v8bf;
typedef __attribute__((ext_vector_type(8)))  float  v8f;
typedef __attribute__((ext_vector_type(4)))  float  v4f;
typedef __attribute__((ext_vector_type(4)))  int    v4i;

#define HID  768
#define NH   12
#define DH   64
#define BB   32
#define SS   512
#define MTOT (BB*SS)        // 16384

// ---------------- CDNA5 async global->LDS ----------------
#if __has_builtin(__builtin_amdgcn_global_load_async_to_lds_b128)
#  define HAVE_ASYNC_LDS 1
#else
#  define HAVE_ASYNC_LDS 0
#endif

typedef __attribute__((address_space(1))) v4i* gas_v4i_ptr;   // global int4*
typedef __attribute__((address_space(3))) v4i* las_v4i_ptr;   // LDS    int4*

// copy 16 bytes global -> LDS
__device__ __forceinline__ void copy16_g2l(bf16_t* ldst, const bf16_t* gsrc) {
#if HAVE_ASYNC_LDS
    __builtin_amdgcn_global_load_async_to_lds_b128(
        (gas_v4i_ptr)(void*)gsrc, (las_v4i_ptr)(void*)ldst, 0, 0);
#else
    *(v8bf*)ldst = *(const v8bf*)gsrc;
#endif
}

__device__ __forceinline__ void wait_async_lds() {
#if HAVE_ASYNC_LDS
#  if __has_builtin(__builtin_amdgcn_s_wait_asynccnt)
    __builtin_amdgcn_s_wait_asynccnt(0);
#  else
    asm volatile("s_wait_asynccnt 0" ::: "memory");
#  endif
#endif
}

// ---------------- helpers ----------------
__device__ __forceinline__ v16bf load_frag(const bf16_t* p0, const bf16_t* p1) {
    v8bf a = *(const v8bf*)p0;
    v8bf b = *(const v8bf*)p1;
    v16bf r;
#pragma unroll
    for (int i = 0; i < 8; ++i) { r[i] = a[i]; r[i + 8] = b[i]; }
    return r;
}

__device__ __forceinline__ v8f wmma_bf16(v16bf a, v16bf b, v8f c) {
    return __builtin_amdgcn_wmma_f32_16x16x32_bf16(false, a, false, b, (short)0, c, false, false);
}

// ---------------- kernel 1: f32 -> bf16 ----------------
__global__ __launch_bounds__(256)
void f32_to_bf16_kernel(const float* __restrict__ src, bf16_t* __restrict__ dst, int n) {
    int i = (blockIdx.x * 256 + threadIdx.x) * 4;
    if (i < n) {
        v4f f = *(const v4f*)(src + i);
        dst[i + 0] = (bf16_t)f[0];
        dst[i + 1] = (bf16_t)f[1];
        dst[i + 2] = (bf16_t)f[2];
        dst[i + 3] = (bf16_t)f[3];
    }
}

// ---------------- kernel 2: fused QKV GEMM ----------------
// Y[m,n] = sum_k X[m,k]*W[n,k] + b[n]   (torch Linear: x @ W.T + b)
// which = blockIdx.z: 0->Q, 1->K (row-major [token, H]), 2->V stored transposed
// Vt[((b*NH+h)*DH+d)*S + s]
#define TM 128
#define TN 128
#define TK 32
#define LPAD 8   // bf16 row stride 40 elems = 80B (16B aligned)

__global__ __launch_bounds__(256)
void qkv_gemm_kernel(const bf16_t* __restrict__ X,
                     const bf16_t* __restrict__ Wq, const bf16_t* __restrict__ Wk,
                     const bf16_t* __restrict__ Wv,
                     const float* __restrict__ bq, const float* __restrict__ bk,
                     const float* __restrict__ bv,
                     bf16_t* __restrict__ Qo, bf16_t* __restrict__ Ko,
                     bf16_t* __restrict__ Vt)
{
    __shared__ bf16_t As[2][TM][TK + LPAD];
    __shared__ bf16_t Bs[2][TN][TK + LPAD];

    const int which = blockIdx.z;
    const bf16_t* W   = (which == 0) ? Wq : (which == 1) ? Wk : Wv;
    const float*  bia = (which == 0) ? bq : (which == 1) ? bk : bv;

    const int m0   = blockIdx.x * TM;
    const int n0   = blockIdx.y * TN;
    const int tid  = threadIdx.x;
    const int lane = tid & 31;
    const int wid  = tid >> 5;
    const int mq   = wid & 3;    // rows  [mq*32, mq*32+32)
    const int nq   = wid >> 2;   // cols  [nq*64, nq*64+64)
    const int l15  = lane & 15;
    const int kh   = lane >> 4;  // 0/1 -> K half

    v8f acc[2][4] = {};

    // each thread moves two 16B chunks per tile per matrix
    const int chunk0 = tid;            // 0..255
    const int chunk1 = tid + 256;      // 256..511
    const int r0 = chunk0 >> 2, c0 = (chunk0 & 3) * 8;
    const int r1 = chunk1 >> 2, c1 = (chunk1 & 3) * 8;

    auto load_tile = [&](int buf, int k0) {
        copy16_g2l(&As[buf][r0][c0], &X[(size_t)(m0 + r0) * HID + k0 + c0]);
        copy16_g2l(&Bs[buf][r0][c0], &W[(size_t)(n0 + r0) * HID + k0 + c0]);
        copy16_g2l(&As[buf][r1][c1], &X[(size_t)(m0 + r1) * HID + k0 + c1]);
        copy16_g2l(&Bs[buf][r1][c1], &W[(size_t)(n0 + r1) * HID + k0 + c1]);
    };

    load_tile(0, 0);
    wait_async_lds();
    __syncthreads();

    int buf = 0;
    for (int k0 = 0; k0 < HID; k0 += TK) {
        if (k0 + TK < HID) {
            load_tile(buf ^ 1, k0 + TK);
            if (k0 + 2 * TK < HID) {   // prefetch the slab after next
                __builtin_prefetch(&X[(size_t)(m0 + r0) * HID + k0 + 2 * TK], 0, 1);
                __builtin_prefetch(&W[(size_t)(n0 + r0) * HID + k0 + 2 * TK], 0, 1);
            }
        }

        v16bf afr[2], bfr[4];
#pragma unroll
        for (int t = 0; t < 2; ++t) {
            const bf16_t* p = &As[buf][mq * 32 + t * 16 + l15][kh * 8];
            afr[t] = load_frag(p, p + 16);
        }
#pragma unroll
        for (int t = 0; t < 4; ++t) {
            const bf16_t* p = &Bs[buf][nq * 64 + t * 16 + l15][kh * 8];
            bfr[t] = load_frag(p, p + 16);
        }
#pragma unroll
        for (int i = 0; i < 2; ++i)
#pragma unroll
            for (int j = 0; j < 4; ++j)
                acc[i][j] = wmma_bf16(afr[i], bfr[j], acc[i][j]);

        wait_async_lds();
        __syncthreads();
        buf ^= 1;
    }

    if (which < 2) {
        bf16_t* O = (which == 0) ? Qo : Ko;
#pragma unroll
        for (int t = 0; t < 2; ++t) {
            int rbase = m0 + mq * 32 + t * 16 + kh * 8;
#pragma unroll
            for (int j = 0; j < 4; ++j) {
                int col = n0 + nq * 64 + j * 16 + l15;
                float bv_ = bia[col];
#pragma unroll
                for (int r = 0; r < 8; ++r)
                    O[(size_t)(rbase + r) * HID + col] = (bf16_t)(acc[t][j][r] + bv_);
            }
        }
    } else {
#pragma unroll
        for (int t = 0; t < 2; ++t) {
            int rbase = m0 + mq * 32 + t * 16 + kh * 8;
            int bidx = rbase >> 9;      // batch
            int s    = rbase & 511;     // token within batch (8 consecutive)
#pragma unroll
            for (int j = 0; j < 4; ++j) {
                int col = n0 + nq * 64 + j * 16 + l15;   // h*64 + d
                int hh = col >> 6, dd = col & 63;
                float bv_ = bia[col];
                v8bf pk;
#pragma unroll
                for (int r = 0; r < 8; ++r) pk[r] = (bf16_t)(acc[t][j][r] + bv_);
                *(v8bf*)&Vt[(((size_t)bidx * NH + hh) * DH + dd) * SS + s] = pk;
            }
        }
    }
}

// ---------------- kernel 3: fused attention (scores in LDS) ----------------
#define SROW 528   // f32 score row stride (512 + 16 pad), 2112B (16B multiple)

__global__ __launch_bounds__(256)
void attention_kernel(const bf16_t* __restrict__ Q, const bf16_t* __restrict__ K,
                      const bf16_t* __restrict__ Vt, float* __restrict__ out)
{
    extern __shared__ float smem[];
    float* sc   = smem;                 // [64][SROW]
    float* red  = smem + 64 * SROW;     // [64][4] partial max
    float* red2 = red + 256;            // [64][4] partial sum

    const int bh = blockIdx.x;
    const int b  = bh / NH;
    const int h  = bh % NH;
    const int q0 = blockIdx.y * 64;

    const int tid  = threadIdx.x;
    const int lane = tid & 31;
    const int wid  = tid >> 5;
    const int m    = wid & 3;      // query sub-tile 0..3
    const int grp  = wid >> 2;     // 0/1
    const int l15  = lane & 15;
    const int kh   = lane >> 4;

    const bf16_t* Qb = Q + (size_t)b * SS * HID + h * DH;
    const bf16_t* Kb = K + (size_t)b * SS * HID + h * DH;
    const bf16_t* Vb = Vt + (size_t)bh * DH * SS;

    // --- load Q fragments (reused across all key tiles) ---
    v16bf qa[2];
    {
        int qrow = q0 + m * 16 + l15;
#pragma unroll
        for (int ks = 0; ks < 2; ++ks) {
            const bf16_t* p = Qb + (size_t)qrow * HID + ks * 32 + kh * 8;
            qa[ks] = load_frag(p, p + 16);
        }
    }

    // --- phase 1: scores = (Q K^T) * 1/sqrt(DH) into LDS ---
    for (int nt = grp * 16; nt < grp * 16 + 16; ++nt) {
        v8f a = {};
        int key = nt * 16 + l15;
        if (nt + 1 < grp * 16 + 16)   // prefetch next key tile row
            __builtin_prefetch(Kb + (size_t)(key + 16) * HID + kh * 8, 0, 1);
#pragma unroll
        for (int ks = 0; ks < 2; ++ks) {
            const bf16_t* p = Kb + (size_t)key * HID + ks * 32 + kh * 8;
            v16bf bf = load_frag(p, p + 16);
            a = wmma_bf16(qa[ks], bf, a);
        }
        int col = nt * 16 + l15;
        int rb  = m * 16 + kh * 8;
#pragma unroll
        for (int r = 0; r < 8; ++r)
            sc[(rb + r) * SROW + col] = a[r] * 0.125f;   // 1/sqrt(64)
    }
    __syncthreads();

    // --- phase 2: softmax over 512 keys, 4 threads per row ---
    {
        int r = tid & 63;
        int c = tid >> 6;                    // 0..3
        float* row = sc + r * SROW + c * 128;
        float mx = -3.0e38f;
#pragma unroll 4
        for (int i = 0; i < 128; ++i) mx = fmaxf(mx, row[i]);
        red[r * 4 + c] = mx;
        __syncthreads();
        float rmax = fmaxf(fmaxf(red[r * 4 + 0], red[r * 4 + 1]),
                           fmaxf(red[r * 4 + 2], red[r * 4 + 3]));
        float sum = 0.f;
#pragma unroll 4
        for (int i = 0; i < 128; ++i) {
            float e = __expf(row[i] - rmax);
            row[i] = e;
            sum += e;
        }
        red2[r * 4 + c] = sum;
        __syncthreads();
        float inv = 1.0f / (red2[r * 4 + 0] + red2[r * 4 + 1] +
                            red2[r * 4 + 2] + red2[r * 4 + 3]);
#pragma unroll 4
        for (int i = 0; i < 128; ++i) row[i] *= inv;
    }
    __syncthreads();

    // --- phase 3: ctx = P @ V (A from LDS probs, B from transposed V) ---
    v8f c2[2] = {};
    const int n0t = grp * 2;
    const int scrow = m * 16 + l15;
    for (int ks = 0; ks < 16; ++ks) {
        int kb = ks * 32 + kh * 8;
        const float* p0 = &sc[scrow * SROW + kb];
        v4f f0 = *(const v4f*)(p0 + 0);
        v4f f1 = *(const v4f*)(p0 + 4);
        v4f f2 = *(const v4f*)(p0 + 16);
        v4f f3 = *(const v4f*)(p0 + 20);
        v16bf a;
#pragma unroll
        for (int i = 0; i < 4; ++i) {
            a[i]      = (bf16_t)f0[i];
            a[4 + i]  = (bf16_t)f1[i];
            a[8 + i]  = (bf16_t)f2[i];
            a[12 + i] = (bf16_t)f3[i];
        }
#pragma unroll
        for (int t = 0; t < 2; ++t) {
            int d = (n0t + t) * 16 + l15;
            const bf16_t* vp = Vb + (size_t)d * SS + ks * 32 + kh * 8;
            v16bf bf = load_frag(vp, vp + 16);
            c2[t] = wmma_bf16(a, bf, c2[t]);
        }
    }

    // --- phase 4: tanh + store f32 ---
#pragma unroll
    for (int t = 0; t < 2; ++t) {
        int dcol  = h * DH + (n0t + t) * 16 + l15;
        int rbase = q0 + m * 16 + kh * 8;
#pragma unroll
        for (int r = 0; r < 8; ++r)
            out[((size_t)b * SS + rbase + r) * HID + dcol] = tanhf(c2[t][r]);
    }
}

// ---------------- host ----------------
extern "C" void kernel_launch(void* const* d_in, const int* in_sizes, int n_in,
                              void* d_out, int out_size, void* d_ws, size_t ws_size,
                              hipStream_t stream) {
    const float* X  = (const float*)d_in[0];
    const float* Wq = (const float*)d_in[1];
    const float* bq = (const float*)d_in[2];
    const float* Wk = (const float*)d_in[3];
    const float* bk = (const float*)d_in[4];
    const float* Wv = (const float*)d_in[5];
    const float* bv = (const float*)d_in[6];
    float* out = (float*)d_out;

    char* ws = (char*)d_ws;
    const size_t XB_SZ = (size_t)MTOT * HID * sizeof(bf16_t);   // 25.2 MB
    const size_t WB_SZ = (size_t)HID * HID * sizeof(bf16_t);    // 1.18 MB
    bf16_t* Xb  = (bf16_t*)(ws);
    bf16_t* Wqb = (bf16_t*)(ws + XB_SZ);
    bf16_t* Wkb = (bf16_t*)(ws + XB_SZ + WB_SZ);
    bf16_t* Wvb = (bf16_t*)(ws + XB_SZ + 2 * WB_SZ);
    bf16_t* Qb  = (bf16_t*)(ws + XB_SZ + 3 * WB_SZ);
    bf16_t* Kb  = (bf16_t*)(ws + 2 * XB_SZ + 3 * WB_SZ);
    bf16_t* Vt  = (bf16_t*)(ws + 3 * XB_SZ + 3 * WB_SZ);

    const int nx = MTOT * HID;   // 12,582,912
    const int nw = HID * HID;    // 589,824
    f32_to_bf16_kernel<<<nx / 1024, 256, 0, stream>>>(X,  Xb,  nx);
    f32_to_bf16_kernel<<<nw / 1024, 256, 0, stream>>>(Wq, Wqb, nw);
    f32_to_bf16_kernel<<<nw / 1024, 256, 0, stream>>>(Wk, Wkb, nw);
    f32_to_bf16_kernel<<<nw / 1024, 256, 0, stream>>>(Wv, Wvb, nw);

    dim3 ggrid(MTOT / TM, HID / TN, 3);   // (128, 6, 3)
    qkv_gemm_kernel<<<ggrid, 256, 0, stream>>>(Xb, Wqb, Wkb, Wvb, bq, bk, bv,
                                               Qb, Kb, Vt);

    size_t smem = (size_t)(64 * SROW + 512) * sizeof(float);   // ~134 KB
    (void)hipFuncSetAttribute((const void*)attention_kernel,
                              hipFuncAttributeMaxDynamicSharedMemorySize, (int)smem);
    attention_kernel<<<dim3(BB * NH, SS / 64), 256, smem, stream>>>(Qb, Kb, Vt, out);
}
